// DeformRoIPoolingPack_6468220748438
// MI455X (gfx1250) — compile-verified, matrix operand order checked by reading, not run on hardware
//
#include <hip/hip_runtime.h>
#include <hip/hip_bf16.h>

#define SCALE   0.0625f
#define PP      7
#define C_DIM   256
#define H_DIM   100
#define W_DIM   100
#define B_DIM   2
#define N_ROIS  512
#define FIN     (PP*PP*C_DIM)   /* 12544 */
#define FC      1024
#define NOFF    147             /* 3*49 */

typedef __attribute__((ext_vector_type(16))) __bf16 v16bf;
typedef __attribute__((ext_vector_type(8)))  float  v8f;

static __device__ __forceinline__ unsigned short f2bf(float f) {
    unsigned int u = __float_as_uint(f);
    u += 0x7FFFu + ((u >> 16) & 1u);          // round-to-nearest-even
    return (unsigned short)(u >> 16);
}
static __device__ __forceinline__ unsigned int pack2(float lo, float hi) {
    return (unsigned int)f2bf(lo) | ((unsigned int)f2bf(hi) << 16);
}

// ---------------- feature transpose: (B,C,H,W) -> (B,H,W,C) ----------------
__global__ void transpose_feat(const float* __restrict__ feat, float* __restrict__ featT) {
    int idx = blockIdx.x * blockDim.x + threadIdx.x;
    const int total = B_DIM * H_DIM * W_DIM * C_DIM;
    if (idx >= total) return;
    int c = idx % C_DIM;
    int w = (idx / C_DIM) % W_DIM;
    int h = (idx / (C_DIM * W_DIM)) % H_DIM;
    int b = idx / (C_DIM * W_DIM * H_DIM);
    featT[idx] = feat[((b * C_DIM + c) * H_DIM + h) * W_DIM + w];
}

// ---------------- deformable PSROI pool (group_size=1, part==pooled) -------
// One block per (roi, bin); one thread per channel (coalesced gathers on featT).
template<bool TRANS>
__global__ void deform_pool(const float* __restrict__ featT,
                            const float* __restrict__ rois,
                            const float* __restrict__ om,      // N x 147 (TRANS only)
                            unsigned short* __restrict__ xout, // bf16 (N x 12544), !TRANS
                            float* __restrict__ fout)          // f32 (N,C,7,7), TRANS
{
    const int blk = blockIdx.x;
    const int n  = blk / (PP * PP);
    const int p  = blk % (PP * PP);
    const int ph = p / PP, pw = p % PP;
    const int c  = threadIdx.x;

    const float r0 = rois[n*5+0], r1 = rois[n*5+1], r2 = rois[n*5+2];
    const float r3 = rois[n*5+3], r4 = rois[n*5+4];
    const int b = (int)r0;
    const float rsw = roundf(r1) * SCALE - 0.5f;
    const float rsh = roundf(r2) * SCALE - 0.5f;
    const float rew = (roundf(r3) + 1.0f) * SCALE - 0.5f;
    const float reh = (roundf(r4) + 1.0f) * SCALE - 0.5f;
    const float roi_w = fmaxf(rew - rsw, 0.1f);
    const float roi_h = fmaxf(reh - rsh, 0.1f);
    const float bin_h = roi_h / PP, bin_w = roi_w / PP;
    const float sub_h = bin_h * 0.25f, sub_w = bin_w * 0.25f;

    float tx = 0.f, ty = 0.f, mask = 1.f;
    if (TRANS) {
        // part_h[ph]==ph, part_w[pw]==pw since PART_SIZE==POOLED
        tx = om[n*NOFF + p] * 0.1f;
        ty = om[n*NOFF + 49 + p] * 0.1f;
        mask = 1.0f / (1.0f + expf(-om[n*NOFF + 98 + p]));
    }
    const float hstart = (float)ph * bin_h + rsh + ty * roi_h;
    const float wstart = (float)pw * bin_w + rsw + tx * roi_w;

    float acc = 0.f;
    int cnt = 0;
    #pragma unroll
    for (int iy = 0; iy < 4; ++iy) {
        #pragma unroll
        for (int ix = 0; ix < 4; ++ix) {
            float h = hstart + iy * sub_h;
            float w = wstart + ix * sub_w;
            bool valid = (w >= -0.5f) && (w <= W_DIM - 0.5f) &&
                         (h >= -0.5f) && (h <= H_DIM - 0.5f);
            if (!valid) continue;                 // uniform across block
            cnt++;
            float hc = fminf(fmaxf(h, 0.f), (float)(H_DIM - 1));
            float wc = fminf(fmaxf(w, 0.f), (float)(W_DIM - 1));
            int h0 = (int)floorf(hc), w0 = (int)floorf(wc);
            int h1 = min(h0 + 1, H_DIM - 1), w1 = min(w0 + 1, W_DIM - 1);
            float lh = hc - (float)h0, lw = wc - (float)w0;
            const float* base = featT + (size_t)b * H_DIM * W_DIM * C_DIM;
            float v00 = base[(h0 * W_DIM + w0) * C_DIM + c];
            float v01 = base[(h0 * W_DIM + w1) * C_DIM + c];
            float v10 = base[(h1 * W_DIM + w0) * C_DIM + c];
            float v11 = base[(h1 * W_DIM + w1) * C_DIM + c];
            acc += v00 * (1.f - lh) * (1.f - lw) + v01 * (1.f - lh) * lw
                 + v10 * lh * (1.f - lw)         + v11 * lh * lw;
        }
    }
    float val = (cnt > 0) ? acc / (float)cnt : 0.f;
    if (TRANS) {
        fout[(size_t)(n * C_DIM + c) * (PP * PP) + p] = val * mask;
    } else {
        xout[(size_t)n * FIN + c * (PP * PP) + p] = f2bf(val);
    }
}

// ---------------- bf16 WMMA GEMM: out = act(A @ Bw^T + bias) ---------------
// A: M x K bf16 (workspace). Bw: Nvalid x K f32 (original weights, converted
// to bf16 in-register during LDS staging).
// 256 threads = 8 waves. Block tile 64(M) x 128(N), K step 64.
// A tile is staged with global_load_async_to_lds_b128 (ASYNCcnt path, no VGPR
// round-trip); B tile goes through VGPRs for the f32->bf16 conversion.
// Each wave owns a 32x32 output tile (2x2 subtiles of 16x16) and issues
// 8 x v_wmma_f32_16x16x32_bf16 per barrier pair, reusing each fragment twice.
#define LDA_STRIDE 36   /* uints per A row: multiple of 4 -> 16B rows for B128 async */
#define LDB_STRIDE 33   /* odd stride -> conflict-free */

template<bool RELU, bool OUTF32, bool NGUARD>
__global__ __launch_bounds__(256) void gemm_wmma(
                          const unsigned short* __restrict__ A,
                          const float* __restrict__ Bw,
                          const float* __restrict__ bias,
                          unsigned short* __restrict__ outBF,
                          float* __restrict__ outF,
                          int K, int Nvalid, int ldOut)
{
    __shared__ unsigned int ldsA[64 * LDA_STRIDE];
    __shared__ unsigned int ldsB[128 * LDB_STRIDE];

    const int t     = threadIdx.x;
    const int lane  = t & 31;
    const int wv    = t >> 5;
    const int wm    = wv >> 2;          // 0..1  (M sub-block of 32)
    const int wn    = wv & 3;           // 0..3  (N sub-block of 32)
    const int row16 = lane & 15;
    const int hi    = lane >> 4;

    const int m0 = blockIdx.y * 64;
    const int n0 = blockIdx.x * 128;

    // staging map: A -> thread covers 8 uints (16 bf16) of one row
    const int ar  = t >> 2;             // 0..63
    const int aq  = (t & 3) * 8;        // uint offset 0,8,16,24
    // staging map: B -> thread covers 32 f32 of one row
    const int br  = t >> 1;             // 0..127
    const int bq  = (t & 1) * 32;       // f32 offset 0 or 32

    // LDS byte offset for this thread's async A destination (flat shared
    // pointer: LDS aperture keeps the byte offset in addr[31:0])
    const unsigned aLds = (unsigned)(size_t)&ldsA[ar * LDA_STRIDE + aq];

    v8f acc[4];
    #pragma unroll
    for (int i = 0; i < 4; ++i) acc[i] = (v8f){0.f,0.f,0.f,0.f,0.f,0.f,0.f,0.f};

    const int kIters = K >> 6;
    for (int it = 0; it < kIters; ++it) {
        const int k0 = it * 64;
        // ---- stage A tile: 64 rows x 64 bf16, async VMEM -> LDS ----
        {
            const unsigned short* gsrc =
                A + (size_t)(m0 + ar) * K + (size_t)k0 + (size_t)aq * 2;
            // INST_OFFSET is added to both the global and the LDS address,
            // so offset:0 / offset:16 covers this thread's 32-byte slice.
            asm volatile(
                "global_load_async_to_lds_b128 %0, %1, off\n\t"
                "global_load_async_to_lds_b128 %0, %1, off offset:16"
                :: "v"(aLds), "v"(gsrc) : "memory");
        }
        // ---- stage B tile: 128 rows x 64 f32 -> bf16 ----
        {
            const int rrow = n0 + br;
            float f[32];
            if (!NGUARD || rrow < Nvalid) {
                const float* src = Bw + (size_t)rrow * K + (size_t)(k0 + bq);
                #pragma unroll
                for (int j = 0; j < 8; ++j) {
                    float4 q = *(const float4*)(src + 4 * j);
                    f[4*j+0] = q.x; f[4*j+1] = q.y; f[4*j+2] = q.z; f[4*j+3] = q.w;
                }
                if (it + 1 < kIters)               // prefetch next K chunk of weights
                    __builtin_prefetch(src + 64, 0, 0);
            } else {
                #pragma unroll
                for (int j = 0; j < 32; ++j) f[j] = 0.f;
            }
            int li = br * LDB_STRIDE + (bq >> 1);
            #pragma unroll
            for (int j = 0; j < 16; ++j)
                ldsB[li + j] = pack2(f[2*j], f[2*j+1]);
        }
        asm volatile("s_wait_asynccnt 0x0" ::: "memory");
        __syncthreads();

        // ---- 2 K sub-chunks x (2x2) subtiles = 8 WMMAs ----
        #pragma unroll
        for (int ks = 0; ks < 2; ++ks) {
            const int kb = ks * 16;                // uint offset within row
            union { unsigned int u[8]; v16bf v; } fa0, fa1, fb0, fb1;
            const int a0 = (wm * 32      + row16) * LDA_STRIDE + kb;
            const int a1 = (wm * 32 + 16 + row16) * LDA_STRIDE + kb;
            const int b0 = (wn * 32      + row16) * LDB_STRIDE + kb;
            const int b1 = (wn * 32 + 16 + row16) * LDB_STRIDE + kb;
            #pragma unroll
            for (int i = 0; i < 8; ++i) {
                // CDNA5 16-bit A/B fragment K-pair index (§7.12.2):
                // i<4 -> hi*4+i ; i>=4 -> 8 + hi*4 + (i-4)
                int off = (i & 3) + hi * 4 + (i >> 2) * 8;
                fa0.u[i] = ldsA[a0 + off];
                fa1.u[i] = ldsA[a1 + off];
                fb0.u[i] = ldsB[b0 + off];
                fb1.u[i] = ldsB[b1 + off];
            }
            acc[0] = __builtin_amdgcn_wmma_f32_16x16x32_bf16(false, fa0.v, false, fb0.v, (short)0, acc[0], false, false);
            acc[1] = __builtin_amdgcn_wmma_f32_16x16x32_bf16(false, fa0.v, false, fb1.v, (short)0, acc[1], false, false);
            acc[2] = __builtin_amdgcn_wmma_f32_16x16x32_bf16(false, fa1.v, false, fb0.v, (short)0, acc[2], false, false);
            acc[3] = __builtin_amdgcn_wmma_f32_16x16x32_bf16(false, fa1.v, false, fb1.v, (short)0, acc[3], false, false);
        }
        __syncthreads();
    }

    // ---- epilogue: D layout M = vgpr + 8*hi, N = lane%16 ----
    #pragma unroll
    for (int sm = 0; sm < 2; ++sm) {
        #pragma unroll
        for (int sn = 0; sn < 2; ++sn) {
            const v8f a = acc[sm * 2 + sn];
            const int col = n0 + wn * 32 + sn * 16 + row16;
            if (!NGUARD || col < Nvalid) {
                const float bv = bias[col];
                #pragma unroll
                for (int r = 0; r < 8; ++r) {
                    int mRow = m0 + wm * 32 + sm * 16 + r + 8 * hi;
                    float v = a[r] + bv;
                    if (RELU) v = fmaxf(v, 0.f);
                    if (OUTF32) outF[(size_t)mRow * ldOut + col] = v;
                    else        outBF[(size_t)mRow * ldOut + col] = f2bf(v);
                }
            }
        }
    }
}

extern "C" void kernel_launch(void* const* d_in, const int* in_sizes, int n_in,
                              void* d_out, int out_size, void* d_ws, size_t ws_size,
                              hipStream_t stream) {
    const float* feat = (const float*)d_in[0];
    const float* rois = (const float*)d_in[1];
    const float* w1   = (const float*)d_in[2];
    const float* b1   = (const float*)d_in[3];
    const float* w2   = (const float*)d_in[4];
    const float* b2   = (const float*)d_in[5];
    const float* w3   = (const float*)d_in[6];
    const float* b3   = (const float*)d_in[7];
    float* out = (float*)d_out;

    char* ws = (char*)d_ws;
    size_t off = 0;
    auto alloc = [&](size_t bytes) -> char* {
        char* p = ws + off;
        off += bytes;
        off = (off + 255) & ~(size_t)255;
        return p;
    };
    float*          featT = (float*)alloc((size_t)B_DIM * H_DIM * W_DIM * C_DIM * 4);
    unsigned short* xbf   = (unsigned short*)alloc((size_t)N_ROIS * FIN * 2);
    unsigned short* y1    = (unsigned short*)alloc((size_t)N_ROIS * FC * 2);
    unsigned short* y2    = (unsigned short*)alloc((size_t)N_ROIS * FC * 2);
    float*          om    = (float*)alloc((size_t)N_ROIS * NOFF * 4);

    const int totalT = B_DIM * H_DIM * W_DIM * C_DIM;
    transpose_feat<<<(totalT + 255) / 256, 256, 0, stream>>>(feat, featT);

    deform_pool<false><<<N_ROIS * PP * PP, C_DIM, 0, stream>>>(
        featT, rois, nullptr, xbf, nullptr);

    gemm_wmma<true, false, false><<<dim3(FC / 128, N_ROIS / 64), 256, 0, stream>>>(
        xbf, w1, b1, y1, nullptr, FIN, FC, FC);

    gemm_wmma<true, false, false><<<dim3(FC / 128, N_ROIS / 64), 256, 0, stream>>>(
        y1, w2, b2, y2, nullptr, FC, FC, FC);

    gemm_wmma<false, true, true><<<dim3(2 /*ceil(147/128)*/, N_ROIS / 64), 256, 0, stream>>>(
        y2, w3, b3, nullptr, om, FC, NOFF, NOFF);

    deform_pool<true><<<N_ROIS * PP * PP, C_DIM, 0, stream>>>(
        featT, rois, om, nullptr, out);
}
